// NCNPredictor_3075196584346
// MI455X (gfx1250) — matrix operand order repaired and checked by reading, not compiled, not run on hardware
//
#include <hip/hip_runtime.h>

// ---------------------------------------------------------------------------
// NCN predictor (NCN_mode==0, k=4) for gfx1250 / MI455X.
//   K0  zero degree array
//   K1  degree count (duplicates kept -> multiplicity semantics)
//   K2  serial exclusive prefix scan -> rowptr, cursor
//   K3  CSR fill (both directions)
//   K4  per-target feature build -> xs in f16 row-major [4096,512]
//   K4b pack W1 -> f16 WMMA B-fragment layout (done once, reused 4096x)
//   K5  h = relu(xs @ W1 + b1): async-to-LDS B staging + v_wmma_f32_16x16x32_f16
//   K6  out = h @ W2 + b2
// ---------------------------------------------------------------------------

typedef __attribute__((ext_vector_type(16))) _Float16 v16h;
typedef __attribute__((ext_vector_type(8)))  _Float16 v8h;
typedef __attribute__((ext_vector_type(8)))  float    v8f;

#define IN_CH   128
#define FEAT    512          // 4 * IN_CH
#define HIDDEN  256
#define MAXJ    1536
#define MAXM    1024
#define KSTEPS  (FEAT / 32)  // 16 WMMA K-steps
#define NTILES  (HIDDEN / 16)

// ---- K0 --------------------------------------------------------------------
__global__ void ncn_zero_i32(int* p, int n) {
    int i = blockIdx.x * blockDim.x + threadIdx.x;
    if (i < n) p[i] = 0;
}

// ---- K1 --------------------------------------------------------------------
__global__ void ncn_degrees(const int* edge, int n_edges, int* deg) {
    int e = blockIdx.x * blockDim.x + threadIdx.x;
    if (e >= n_edges) return;
    atomicAdd(&deg[edge[e]], 1);
    atomicAdd(&deg[edge[n_edges + e]], 1);
}

// ---- K2 --------------------------------------------------------------------
__global__ void ncn_scan(const int* deg, int n_nodes, int* rowptr, int* cursor) {
    if (blockIdx.x != 0 || threadIdx.x != 0) return;
    int run = 0;
    for (int i = 0; i < n_nodes; ++i) {
        rowptr[i] = run;
        cursor[i] = run;
        run += deg[i];
    }
    rowptr[n_nodes] = run;
}

// ---- K3 --------------------------------------------------------------------
__global__ void ncn_fill(const int* edge, int n_edges, int* cursor, int* colidx) {
    int e = blockIdx.x * blockDim.x + threadIdx.x;
    if (e >= n_edges) return;
    int s = edge[e];
    int d = edge[n_edges + e];
    colidx[atomicAdd(&cursor[s], 1)] = d;
    colidx[atomicAdd(&cursor[d], 1)] = s;
}

// ---- K4: feature rows, written as f16 --------------------------------------
__global__ __launch_bounds__(128) void ncn_build_xs(
    const float* __restrict__ x, const int* __restrict__ tar, int n_tar,
    const int* __restrict__ rowptr, const int* __restrict__ colidx,
    _Float16* __restrict__ xsh) {
    int t   = blockIdx.x;
    int tid = threadIdx.x;       // channel 0..127
    int ti  = tar[t];
    int tj  = tar[n_tar + t];

    __shared__ int s_adjj[MAXJ];
    __shared__ int s_match[MAXM];
    __shared__ int s_nj, s_nm, s_acnt;

    if (tid == 0) {
        s_nm = 0; s_acnt = 0;
        int dj = rowptr[tj + 1] - rowptr[tj];
        s_nj = dj < MAXJ ? dj : MAXJ;
    }
    __syncthreads();

    int jstart = rowptr[tj];
    for (int k = tid; k < s_nj; k += 128) s_adjj[k] = colidx[jstart + k];
    __syncthreads();

    int istart = rowptr[ti], iend = rowptr[ti + 1];
    int nj = s_nj;
    for (int k = istart + tid; k < iend; k += 128) {
        int n = colidx[k];
        if (n == tj) atomicAdd(&s_acnt, 1);
        for (int m = 0; m < nj; ++m) {
            if (s_adjj[m] == n) {
                int pos = atomicAdd(&s_nm, 1);
                if (pos < MAXM) s_match[pos] = n;
            }
        }
    }
    __syncthreads();

    float a  = (float)s_acnt;                  // A[ti,tj] == A[tj,ti]
    int   nm = s_nm < MAXM ? s_nm : MAXM;

    float xi = x[ti * IN_CH + tid];
    float xj = x[tj * IN_CH + tid];
    float acc = 0.0f;
    for (int m = 0; m < nm; ++m) acc += x[s_match[m] * IN_CH + tid];

    _Float16* row = xsh + (size_t)t * FEAT;
    row[tid]           = (_Float16)(xi * xj);
    row[IN_CH   + tid] = (_Float16)(a * xi);
    row[2*IN_CH + tid] = (_Float16)(a * xj);
    row[3*IN_CH + tid] = (_Float16)acc;
}

// ---- K4b: pack W1 into f16 B-fragment layout -------------------------------
// pack[((nt*KSTEPS + kb)*32 + lane)*16 + e]:
//   lane<16 : N = nt*16+lane,  K = kb*32 + {0..7, 16..23}[e]
//   lane>=16: N = nt*16+lane-16, K = kb*32 + {8..15, 24..31}[e]
__global__ __launch_bounds__(256) void ncn_pack_w1(
    const float* __restrict__ W1, _Float16* __restrict__ pack) {
    int gid  = blockIdx.x * blockDim.x + threadIdx.x;  // 0..8191
    int lane = gid & 31;
    int kb   = (gid >> 5) & (KSTEPS - 1);
    int nt   = gid >> 9;
    bool hi  = lane >= 16;
    int n    = nt * 16 + (lane & 15);
    _Float16* dst = pack + ((size_t)(nt * KSTEPS + kb) * 32 + lane) * 16;
    #pragma unroll
    for (int v = 0; v < 8; ++v) {
        int k = kb * 32 + 2 * v + (v >= 4 ? 8 : 0) + (hi ? 8 : 0);
        dst[2 * v]     = (_Float16)W1[(size_t)k       * HIDDEN + n];
        dst[2 * v + 1] = (_Float16)W1[(size_t)(k + 1) * HIDDEN + n];
    }
}

// ---- K5: h = relu(xs @ W1 + b1) --------------------------------------------
// Block = 256 threads = 8 waves, all sharing one N-tile's B data.
// B tile (16 KB) staged to LDS via global_load_async_to_lds_b128 (ASYNCcnt),
// consumed via ds_load_b128. Each wave computes one 16x16 output tile per
// grid.x slot: rows = (blockIdx.x*8 + wave)*16.
__global__ __launch_bounds__(256) void ncn_mlp1_wmma(
    const _Float16* __restrict__ xsh, const _Float16* __restrict__ pack,
    const float* __restrict__ b1, float* __restrict__ h) {
    __shared__ _Float16 s_b[KSTEPS * 32 * 16];   // 16 KB

    int tid  = threadIdx.x;
    int lane = tid & 31;
    int wave = tid >> 5;
    int nt   = blockIdx.y;

    // --- async-stage B tile: wave w loads K-steps {2w, 2w+1} ---
    #pragma unroll
    for (int q = 0; q < 2; ++q) {
        int kb = wave * 2 + q;
        const _Float16* src = pack + ((size_t)(nt * KSTEPS + kb) * 32 + lane) * 16;
        unsigned           lds0 = (unsigned)(unsigned long long)&s_b[((kb * 32) + lane) * 16];
        unsigned long long g0   = (unsigned long long)src;
        asm volatile("global_load_async_to_lds_b128 %0, %1, off"
                     :: "v"(lds0), "v"(g0) : "memory");
        unsigned           lds1 = lds0 + 16;
        unsigned long long g1   = g0 + 16;
        asm volatile("global_load_async_to_lds_b128 %0, %1, off"
                     :: "v"(lds1), "v"(g1) : "memory");
    }
    asm volatile("s_wait_asynccnt 0x0" ::: "memory");
    __syncthreads();

    int  lo16 = lane & 15;
    bool hi   = lane >= 16;
    int  rowA = (blockIdx.x * 8 + wave) * 16 + lo16;
    const _Float16* arow = xsh + (size_t)rowA * FEAT;
    __builtin_prefetch(arow, 0, 0);   // global_prefetch

    v8f c = {};
    #pragma unroll
    for (int kb = 0; kb < KSTEPS; ++kb) {
        // A 16x32 fragment: lane-lo K={kb*32+0..7, kb*32+16..23}; lane-hi +8
        v8h a0 = *(const v8h*)(arow + kb * 32 +      (hi ? 8 : 0));
        v8h a1 = *(const v8h*)(arow + kb * 32 + 16 + (hi ? 8 : 0));
        v16h a, b;
        #pragma unroll
        for (int e = 0; e < 8; ++e) { a[e] = a0[e]; a[8 + e] = a1[e]; }
        // B 32x16 fragment from LDS (contiguous 32 B per lane)
        const _Float16* bp = &s_b[((kb * 32) + lane) * 16];
        v8h b0 = *(const v8h*)bp;
        v8h b1v = *(const v8h*)(bp + 8);
        #pragma unroll
        for (int e = 0; e < 8; ++e) { b[e] = b0[e]; b[8 + e] = b1v[e]; }
        c = __builtin_amdgcn_wmma_f32_16x16x32_f16(
                /*neg_a=*/false, a, /*neg_b=*/false, b,
                /*c_mod=*/(short)0, c, /*reuse_a=*/false, /*reuse_b=*/false);
    }

    // D layout: VGPR r -> (M = mbase + r, N = nt*16 + lo16)
    int   n     = nt * 16 + lo16;
    int   mbase = (blockIdx.x * 8 + wave) * 16 + (hi ? 8 : 0);
    float bias  = b1[n];
    #pragma unroll
    for (int r = 0; r < 8; ++r) {
        float val = c[r] + bias;
        h[(size_t)(mbase + r) * HIDDEN + n] = fmaxf(val, 0.0f);
    }
}

// ---- K6 --------------------------------------------------------------------
__global__ __launch_bounds__(256) void ncn_mlp2(
    const float* __restrict__ h, const float* __restrict__ W2,
    const float* __restrict__ b2, float* __restrict__ out) {
    __shared__ float red[HIDDEN];
    int t = blockIdx.x;
    int c = threadIdx.x;
    red[c] = h[(size_t)t * HIDDEN + c] * W2[c];
    __syncthreads();
    for (int s = HIDDEN / 2; s > 0; s >>= 1) {
        if (c < s) red[c] += red[c + s];
        __syncthreads();
    }
    if (c == 0) out[t] = red[0] + b2[0];
}

// ---------------------------------------------------------------------------
extern "C" void kernel_launch(void* const* d_in, const int* in_sizes, int n_in,
                              void* d_out, int out_size, void* d_ws, size_t ws_size,
                              hipStream_t stream) {
    const float* x    = (const float*)d_in[0];
    const int*   edge = (const int*)d_in[1];
    const int*   tar  = (const int*)d_in[2];
    // d_in[3] = NCN_mode (ignored; mode 0 path)
    const float* W1   = (const float*)d_in[4];
    const float* b1   = (const float*)d_in[5];
    const float* W2   = (const float*)d_in[6];
    const float* b2   = (const float*)d_in[7];
    float* out        = (float*)d_out;

    const int n_nodes = in_sizes[0] / IN_CH;      // 10000
    const int n_edges = in_sizes[1] / 2;          // 320000
    const int n_tar   = in_sizes[2] / 2;          // 4096

    // Workspace carve-up.
    int*      deg    = (int*)d_ws;                          // n_nodes
    int*      rowptr = deg + 10016;                         // n_nodes + 1
    int*      cursor = rowptr + 10016;                      // n_nodes
    int*      colidx = cursor + 10016;                      // 2 * n_edges
    _Float16* pack   = (_Float16*)(colidx + 2 * 320000);    // 512*256 f16 (256 KB)
    _Float16* xsh    = pack + (size_t)FEAT * HIDDEN;        // 4096*512 f16 (4 MB)
    float*    h      = (float*)(xsh + (size_t)4096 * FEAT); // 4096*256 f32 (4 MB)

    ncn_zero_i32<<<(n_nodes + 255) / 256, 256, 0, stream>>>(deg, n_nodes);
    ncn_degrees<<<(n_edges + 255) / 256, 256, 0, stream>>>(edge, n_edges, deg);
    ncn_scan<<<1, 1, 0, stream>>>(deg, n_nodes, rowptr, cursor);
    ncn_fill<<<(n_edges + 255) / 256, 256, 0, stream>>>(edge, n_edges, cursor, colidx);
    ncn_build_xs<<<n_tar, 128, 0, stream>>>(x, tar, n_tar, rowptr, colidx, xsh);
    ncn_pack_w1<<<(NTILES * KSTEPS * 32) / 256, 256, 0, stream>>>(W1, pack);
    // 256 row tiles / 8 waves per block = 32 blocks in x; 16 col tiles in y.
    dim3 g5((n_tar / 16) / 8, NTILES);
    ncn_mlp1_wmma<<<g5, 256, 0, stream>>>(xsh, pack, b1, h);
    ncn_mlp2<<<n_tar, HIDDEN, 0, stream>>>(h, W2, b2, out);
}
